// Decoder_11665131176286
// MI455X (gfx1250) — matrix operand order repaired and checked by reading, not compiled, not run on hardware
//
#include <hip/hip_runtime.h>
#include <cstdint>
#include <cstddef>

typedef __attribute__((ext_vector_type(16))) _Float16 v16h;
typedef __attribute__((ext_vector_type(8)))  float    v8f;

#define BB      32
#define T_DEC   400
#define ENC_T   512
#define MEM_D   512
#define ATT_RNN 1024
#define DEC_RNN 1024
#define PRE_D   256
#define ATT_D   128
#define LOC_F   32
#define LOC_K   31
#define N_MEL   80
#define XPAD_K  96      // N_MEL padded up to a multiple of 32
#define NTJ     4       // N-tiles per wave in the GEMM

// ---------------------------------------------------------------------------
// Pack a row-major f32 weight [K,N] into WMMA-B f16 tile layout.
// Tile = 32(K) x 16(N). Per tile: 32 lane-slots x 16 f16 (one v16h load/lane).
// B layout (16-bit, 32x16): lanes 0-15 -> column N=lane, K {0..7,16..23};
// lanes 16-31 -> column N=lane-16, K {8..15,24..31}. Zero-pads K to 32 / N to 16.
// ---------------------------------------------------------------------------
__global__ void pack_b_f16(const float* __restrict__ W, _Float16* __restrict__ Bp,
                           int K, int N, int NT) {
    int nt   = blockIdx.x;
    int kt   = blockIdx.y;
    int lane = threadIdx.x;
    _Float16* dst = Bp + (((size_t)kt * NT + nt) * 32 + lane) * 16;
    int n = nt * 16 + (lane & 15);
#pragma unroll
    for (int e = 0; e < 16; ++e) {
        int k = kt * 32 + ((lane >> 4) << 3) + (e & 7) + ((e >> 3) << 4);
        float v = (k < K && n < N) ? W[(size_t)k * N + n] : 0.0f;
        dst[e] = (_Float16)v;
    }
}

// ---------------------------------------------------------------------------
// Load + convert one 16-row x 32-col A fragment (f32 row-major -> f16 regs)
// per the 16-bit A-matrix VGPR layout (lanes 0-15: K {0..7,16..23},
// lanes 16-31: K {8..15,24..31}); arow already lane-adjusted.
// ---------------------------------------------------------------------------
__device__ __forceinline__ v16h load_a_frag(const float* arow, int k0) {
    float4 a0 = *(const float4*)(arow + k0);
    float4 a1 = *(const float4*)(arow + k0 + 4);
    float4 a2 = *(const float4*)(arow + k0 + 16);
    float4 a3 = *(const float4*)(arow + k0 + 20);
    float af[16] = { a0.x, a0.y, a0.z, a0.w, a1.x, a1.y, a1.z, a1.w,
                     a2.x, a2.y, a2.z, a2.w, a3.x, a3.y, a3.z, a3.w };
    v16h a;
#pragma unroll
    for (int e = 0; e < 16; ++e) a[e] = (_Float16)af[e];
    return a;
}

// ---------------------------------------------------------------------------
// WMMA GEMM: C[M,N] (+)= relu?( A[M,K] * Bpacked[K,N] + bias )
// REQUIRES: M % 16 == 0, N % 16 == 0, K % 32 == 0, A rows 16B-aligned.
// One wave computes a 16x(16*NTJ) strip: one A fragment feeds NTJ WMMAs.
// Tile decomposition is scalarized via readfirstlane; the full-strip case
// (jmax==NTJ, the common path) is completely branch-free in the K loop:
// all NTJ B fragments are loaded, then NTJ WMMAs issue back-to-back.
// flags: bit0 = accumulate into C, bit1 = relu.
// ---------------------------------------------------------------------------
__global__ void wmma_gemm(const float* __restrict__ A, const _Float16* __restrict__ Bp,
                          float* __restrict__ C, const float* __restrict__ bias,
                          int M, int N, int K, int lda, int ldc, int flags) {
    int lane = threadIdx.x & 31;
    // wave id is uniform across the wave -> force into SGPR
    int wave = __builtin_amdgcn_readfirstlane((int)(threadIdx.x >> 5));
    int NT   = N >> 4;
    int MT   = M >> 4;
    int KT   = K >> 5;
    int NTG  = (NT + NTJ - 1) / NTJ;

    int tile = (int)blockIdx.x * 4 + wave;         // SGPR
    if (tile >= MT * NTG) return;                  // scalar branch
    int mt   = tile / NTG;
    int nt0  = (tile % NTG) * NTJ;
    int jmax = NT - nt0; if (jmax > NTJ) jmax = NTJ;   // SGPR

    int row = mt * 16 + (lane & 15);
    const float* arow = A + (size_t)row * lda + ((lane >> 4) << 3);

    v8f acc[NTJ];
#pragma unroll
    for (int j = 0; j < NTJ; ++j) acc[j] = (v8f){};

    if (jmax == NTJ) {
        // ---- common path: branch-free K loop ----
        for (int kt = 0; kt < KT; ++kt) {
            v16h a = load_a_frag(arow, kt * 32);
            const _Float16* bbase = Bp + (((size_t)kt * NT + nt0) * 32 + lane) * 16;
            v16h b0 = *(const v16h*)(bbase);
            v16h b1 = *(const v16h*)(bbase + 1 * 32 * 16);
            v16h b2 = *(const v16h*)(bbase + 2 * 32 * 16);
            v16h b3 = *(const v16h*)(bbase + 3 * 32 * 16);
            acc[0] = __builtin_amdgcn_wmma_f32_16x16x32_f16(false, a, false, b0,
                                                            (short)0, acc[0], false, false);
            acc[1] = __builtin_amdgcn_wmma_f32_16x16x32_f16(false, a, false, b1,
                                                            (short)0, acc[1], false, false);
            acc[2] = __builtin_amdgcn_wmma_f32_16x16x32_f16(false, a, false, b2,
                                                            (short)0, acc[2], false, false);
            acc[3] = __builtin_amdgcn_wmma_f32_16x16x32_f16(false, a, false, b3,
                                                            (short)0, acc[3], false, false);
        }
    } else {
        // ---- tail path (only the last tile group when NT % NTJ != 0) ----
        for (int kt = 0; kt < KT; ++kt) {
            v16h a = load_a_frag(arow, kt * 32);
            const _Float16* bbase = Bp + (((size_t)kt * NT + nt0) * 32 + lane) * 16;
#pragma unroll
            for (int j = 0; j < NTJ; ++j) {
                if (j < jmax) {                    // scalar (SGPR) condition
                    v16h b = *(const v16h*)(bbase + (size_t)j * 32 * 16);
                    acc[j] = __builtin_amdgcn_wmma_f32_16x16x32_f16(
                        false, a, false, b, (short)0, acc[j], false, false);
                }
            }
        }
    }

    // D layout: lane&15 -> column; VGPR r -> row m = 8*(lane>=16) + r
    int mbase = mt * 16 + ((lane >> 4) << 3);
#pragma unroll
    for (int j = 0; j < NTJ; ++j) {
        if (j < jmax) {                            // scalar condition
            int n = (nt0 + j) * 16 + (lane & 15);
#pragma unroll
            for (int r = 0; r < 8; ++r) {
                float v = acc[j][r];
                if (bias)      v += bias[n];
                size_t idx = (size_t)(mbase + r) * ldc + n;
                if (flags & 1) v += C[idx];
                if (flags & 2) v = fmaxf(v, 0.0f);
                C[idx] = v;
            }
        }
    }
}

// ---------------------------------------------------------------------------
// xpad[b, 0:96] = { x_t[b, 0:80], 0... }  (guard-free prenet GEMM input)
// ---------------------------------------------------------------------------
__global__ void pad_x(const float* __restrict__ din, float* __restrict__ xpad, int t) {
    int id = blockIdx.x * blockDim.x + threadIdx.x;
    if (id >= BB * XPAD_K) return;
    int b = id / XPAD_K, j = id % XPAD_K;
    xpad[id] = (j < N_MEL) ? din[(size_t)b * T_DEC * N_MEL + (size_t)t * N_MEL + j] : 0.0f;
}

// ---------------------------------------------------------------------------
// dst[b, :] = concat(a[b, 0:da], c[b, 0:db])  for b in [0,BB)
// ---------------------------------------------------------------------------
__global__ void concat2(const float* __restrict__ a, int da,
                        const float* __restrict__ c, int db,
                        float* __restrict__ dst) {
    int D  = da + db;
    int id = blockIdx.x * blockDim.x + threadIdx.x;
    if (id >= BB * D) return;
    int b = id / D, j = id % D;
    dst[id] = (j < da) ? a[(size_t)b * da + j] : c[(size_t)b * db + (j - da)];
}

// ---------------------------------------------------------------------------
// LSTM cell pointwise: gates [BB,4H] (i,f,g,o), h/c updated in place.
// ---------------------------------------------------------------------------
__global__ void lstm_cell(const float* __restrict__ gates, const float* __restrict__ bias,
                          float* __restrict__ h, float* __restrict__ c, int H) {
    int id = blockIdx.x * blockDim.x + threadIdx.x;
    if (id >= BB * H) return;
    int b = id / H, j = id % H;
    const float* g = gates + (size_t)b * 4 * H;
    float gi = g[j]         + bias[j];
    float gf = g[H + j]     + bias[H + j];
    float gg = g[2 * H + j] + bias[2 * H + j];
    float go = g[3 * H + j] + bias[3 * H + j];
    float si = 1.0f / (1.0f + __expf(-gi));
    float sf = 1.0f / (1.0f + __expf(-gf));
    float so = 1.0f / (1.0f + __expf(-go));
    float cn = sf * c[id] + si * tanhf(gg);
    c[id] = cn;
    h[id] = so * tanhf(cn);
}

// ---------------------------------------------------------------------------
// Fused location-sensitive attention. One block per batch element, 512 threads
// (thread = encoder position t for conv/energy/softmax, then context dim d).
// Halo-padded LDS (+/-16 zeros) makes the 31-tap conv branch-free.
// ---------------------------------------------------------------------------
__global__ void attention_step(float* __restrict__ aw, float* __restrict__ awc,
                               float* __restrict__ ctx, const float* __restrict__ q,
                               const float* __restrict__ pmem,
                               const float* __restrict__ memory,
                               const int* __restrict__ mlen,
                               const float* __restrict__ Wconv,   // [LOC_F,2,LOC_K]
                               const float* __restrict__ Wld,     // [LOC_F,ATT_D]
                               const float* __restrict__ v) {     // [ATT_D]
    __shared__ float aw_s[ENC_T + 32];
    __shared__ float awc_s[ENC_T + 32];
    __shared__ float red_s[ENC_T];
    __shared__ float Wld_s[LOC_F * ATT_D];
    __shared__ float v_s[ATT_D];
    __shared__ float Wc_s[LOC_F * 2 * LOC_K];

    int b = blockIdx.x;
    int t = threadIdx.x;

    if (t < 16) {
        aw_s[t] = 0.0f;  awc_s[t] = 0.0f;
        aw_s[ENC_T + 16 + t] = 0.0f;  awc_s[ENC_T + 16 + t] = 0.0f;
    }
    aw_s[16 + t]  = aw[(size_t)b * ENC_T + t];
    awc_s[16 + t] = awc[(size_t)b * ENC_T + t];
    for (int i = t; i < LOC_F * ATT_D; i += ENC_T) Wld_s[i] = Wld[i];
    for (int i = t; i < LOC_F * 2 * LOC_K; i += ENC_T) Wc_s[i] = Wconv[i];
    if (t < ATT_D) v_s[t] = v[t];
    __syncthreads();

    // conv over 2 channels, kernel 31, pad 15 (branch-free via halo)
    float loc[LOC_F];
#pragma unroll 4
    for (int f = 0; f < LOC_F; ++f) {
        float s = 0.0f;
        for (int k = 0; k < LOC_K; ++k) {
            s += aw_s[t + k + 1]  * Wc_s[f * 2 * LOC_K + k];
            s += awc_s[t + k + 1] * Wc_s[f * 2 * LOC_K + LOC_K + k];
        }
        loc[f] = s;
    }

    // energy: e_t = v . tanh(loc_t @ Wld + pmem[b,t] + q[b])
    const float* pm = pmem + ((size_t)b * ENC_T + t) * ATT_D;
    const float* qb = q + (size_t)b * ATT_D;
    float e = 0.0f;
    for (int a = 0; a < ATT_D; ++a) {
        float s = qb[a] + pm[a];
#pragma unroll 8
        for (int f = 0; f < LOC_F; ++f) s += loc[f] * Wld_s[f * ATT_D + a];
        e += v_s[a] * tanhf(s);
    }
    if (t >= mlen[b]) e = -1e9f;

    // masked softmax (block-wide max + sum)
    red_s[t] = e;
    __syncthreads();
    for (int off = ENC_T / 2; off > 0; off >>= 1) {
        if (t < off) red_s[t] = fmaxf(red_s[t], red_s[t + off]);
        __syncthreads();
    }
    float mx = red_s[0];
    __syncthreads();
    float p = __expf(e - mx);
    red_s[t] = p;
    __syncthreads();
    for (int off = ENC_T / 2; off > 0; off >>= 1) {
        if (t < off) red_s[t] += red_s[t + off];
        __syncthreads();
    }
    float awn = p / red_s[0];
    __syncthreads();

    aw_s[16 + t] = awn;
    aw[(size_t)b * ENC_T + t]  = awn;
    awc[(size_t)b * ENC_T + t] = awc_s[16 + t] + awn;
    __syncthreads();

    // ctx[d] = sum_t aw[t] * memory[b, t, d]   (thread = d; coalesced over d)
    float s = 0.0f;
    const float* mb = memory + (size_t)b * ENC_T * MEM_D + t;
    for (int tt = 0; tt < ENC_T; ++tt) s += aw_s[16 + tt] * mb[(size_t)tt * MEM_D];
    ctx[(size_t)b * MEM_D + t] = s;
}

// ---------------------------------------------------------------------------
// Host-side driver
// ---------------------------------------------------------------------------
extern "C" void kernel_launch(void* const* d_in, const int* in_sizes, int n_in,
                              void* d_out, int out_size, void* d_ws, size_t ws_size,
                              hipStream_t stream) {
    (void)in_sizes; (void)n_in; (void)out_size; (void)ws_size;

    const float* memory    = (const float*)d_in[0];   // [32,512,512]
    const float* dec_in    = (const float*)d_in[1];   // [32,400,80]
    const int*   mlen      = (const int*)  d_in[2];   // [32]
    const float* Wp1       = (const float*)d_in[3];   // [80,256]
    const float* Wp2       = (const float*)d_in[4];   // [256,256]
    const float* Wih_att   = (const float*)d_in[5];   // [768,4096]
    const float* Whh_att   = (const float*)d_in[6];   // [1024,4096]
    const float* b_att     = (const float*)d_in[7];   // [4096]
    const float* Wq        = (const float*)d_in[8];   // [1024,128]
    const float* Wmem      = (const float*)d_in[9];   // [512,128]
    const float* vvec      = (const float*)d_in[10];  // [128]
    const float* Wloc_conv = (const float*)d_in[11];  // [32,2,31]
    const float* Wloc_den  = (const float*)d_in[12];  // [32,128]
    const float* Wih_dec   = (const float*)d_in[13];  // [1536,4096]
    const float* Whh_dec   = (const float*)d_in[14];  // [1024,4096]
    const float* b_dec     = (const float*)d_in[15];  // [4096]
    const float* Wproj     = (const float*)d_in[16];  // [1536,80]
    const float* bproj     = (const float*)d_in[17];  // [80]
    float*       out       = (float*)d_out;           // [32,400,80]

    // ---- workspace carve ----
    size_t off = 0;
    char*  base = (char*)d_ws;
    auto carve = [&](size_t bytes) -> char* {
        char* p = base + off;
        off += (bytes + 255) & ~(size_t)255;
        return p;
    };
    auto packedBytes = [](int K, int N) -> size_t {
        return (size_t)((K + 31) / 32) * ((N + 15) / 16) * 512 * sizeof(_Float16);
    };

    _Float16* pWp1  = (_Float16*)carve(packedBytes(80, 256));
    _Float16* pWp2  = (_Float16*)carve(packedBytes(256, 256));
    _Float16* pWia  = (_Float16*)carve(packedBytes(768, 4096));
    _Float16* pWha  = (_Float16*)carve(packedBytes(1024, 4096));
    _Float16* pWq   = (_Float16*)carve(packedBytes(1024, 128));
    _Float16* pWm   = (_Float16*)carve(packedBytes(512, 128));
    _Float16* pWid  = (_Float16*)carve(packedBytes(1536, 4096));
    _Float16* pWhd  = (_Float16*)carve(packedBytes(1024, 4096));
    _Float16* pWpr  = (_Float16*)carve(packedBytes(1536, 80));

    float* pmem      = (float*)carve((size_t)BB * ENC_T * ATT_D * 4); // [B*512,128]
    float* xpad      = (float*)carve((size_t)BB * XPAD_K * 4);
    float* pre1      = (float*)carve((size_t)BB * PRE_D * 4);
    float* pre       = (float*)carve((size_t)BB * PRE_D * 4);
    float* xcat_att  = (float*)carve((size_t)BB * (PRE_D + MEM_D) * 4);
    float* gates_att = (float*)carve((size_t)BB * 4 * ATT_RNN * 4);
    float* att_h     = (float*)carve((size_t)BB * ATT_RNN * 4);
    float* att_c     = (float*)carve((size_t)BB * ATT_RNN * 4);
    float* qbuf      = (float*)carve((size_t)BB * ATT_D * 4);
    float* aw        = (float*)carve((size_t)BB * ENC_T * 4);
    float* awc       = (float*)carve((size_t)BB * ENC_T * 4);
    float* ctx       = (float*)carve((size_t)BB * MEM_D * 4);
    float* xcat_dec  = (float*)carve((size_t)BB * (ATT_RNN + MEM_D) * 4);
    float* gates_dec = (float*)carve((size_t)BB * 4 * DEC_RNN * 4);
    float* dec_h     = (float*)carve((size_t)BB * DEC_RNN * 4);
    float* dec_c     = (float*)carve((size_t)BB * DEC_RNN * 4);
    float* xcat_prj  = (float*)carve((size_t)BB * (DEC_RNN + MEM_D) * 4);

    // ---- helpers ----
    auto pack = [&](const float* W, _Float16* Bp, int K, int N) {
        dim3 g((N + 15) / 16, (K + 31) / 32);
        hipLaunchKernelGGL(pack_b_f16, g, dim3(32), 0, stream, W, Bp, K, N, (N + 15) / 16);
    };
    auto gemm = [&](const float* A, const _Float16* Bp, float* C, const float* bias,
                    int M, int N, int K, int lda, int ldc, int flags) {
        int tiles = (M / 16) * ((N / 16 + NTJ - 1) / NTJ);
        hipLaunchKernelGGL(wmma_gemm, dim3((tiles + 3) / 4), dim3(128), 0, stream,
                           A, Bp, C, bias, M, N, K, lda, ldc, flags);
    };
    auto cat = [&](const float* a, int da, const float* c, int db, float* dst) {
        int tot = BB * (da + db);
        hipLaunchKernelGGL(concat2, dim3((tot + 255) / 256), dim3(256), 0, stream,
                           a, da, c, db, dst);
    };

    // ---- one-time per launch: pack weights to f16 WMMA-B layout ----
    pack(Wp1,     pWp1, 80,   256);
    pack(Wp2,     pWp2, 256,  256);
    pack(Wih_att, pWia, 768,  4096);
    pack(Whh_att, pWha, 1024, 4096);
    pack(Wq,      pWq,  1024, 128);
    pack(Wmem,    pWm,  512,  128);
    pack(Wih_dec, pWid, 1536, 4096);
    pack(Whh_dec, pWhd, 1024, 4096);
    pack(Wproj,   pWpr, 1536, 80);

    // processed_memory = memory @ Wmem : [B*ENC_T, ATT_D]
    gemm(memory, pWm, pmem, nullptr, BB * ENC_T, ATT_D, MEM_D, MEM_D, ATT_D, 0);

    // zero-init recurrent state
    hipMemsetAsync(att_h, 0, (size_t)BB * ATT_RNN * 4, stream);
    hipMemsetAsync(att_c, 0, (size_t)BB * ATT_RNN * 4, stream);
    hipMemsetAsync(dec_h, 0, (size_t)BB * DEC_RNN * 4, stream);
    hipMemsetAsync(dec_c, 0, (size_t)BB * DEC_RNN * 4, stream);
    hipMemsetAsync(aw,    0, (size_t)BB * ENC_T * 4, stream);
    hipMemsetAsync(awc,   0, (size_t)BB * ENC_T * 4, stream);
    hipMemsetAsync(ctx,   0, (size_t)BB * MEM_D * 4, stream);

    // ---- sequential decode: 400 steps, recurrence carried by stream order ----
    for (int t = 0; t < T_DEC; ++t) {
        // stage x_t into zero-padded [32,96] so the prenet GEMM is guard-free
        hipLaunchKernelGGL(pad_x, dim3((BB * XPAD_K + 255) / 256), dim3(256), 0,
                           stream, dec_in, xpad, t);

        // prenet: relu(relu(x@Wp1)@Wp2)
        gemm(xpad, pWp1, pre1, nullptr, BB, PRE_D, XPAD_K, XPAD_K, PRE_D, 2);
        gemm(pre1, pWp2, pre,  nullptr, BB, PRE_D, PRE_D,  PRE_D,  PRE_D, 2);

        // attention LSTM
        cat(pre, PRE_D, ctx, MEM_D, xcat_att);
        gemm(xcat_att, pWia, gates_att, nullptr, BB, 4 * ATT_RNN, PRE_D + MEM_D,
             PRE_D + MEM_D, 4 * ATT_RNN, 0);
        gemm(att_h,    pWha, gates_att, nullptr, BB, 4 * ATT_RNN, ATT_RNN,
             ATT_RNN, 4 * ATT_RNN, 1);
        hipLaunchKernelGGL(lstm_cell, dim3((BB * ATT_RNN + 255) / 256), dim3(256), 0,
                           stream, gates_att, b_att, att_h, att_c, ATT_RNN);

        // location-sensitive attention
        gemm(att_h, pWq, qbuf, nullptr, BB, ATT_D, ATT_RNN, ATT_RNN, ATT_D, 0);
        hipLaunchKernelGGL(attention_step, dim3(BB), dim3(ENC_T), 0, stream,
                           aw, awc, ctx, qbuf, pmem, memory, mlen,
                           Wloc_conv, Wloc_den, vvec);

        // decoder LSTM
        cat(att_h, ATT_RNN, ctx, MEM_D, xcat_dec);
        gemm(xcat_dec, pWid, gates_dec, nullptr, BB, 4 * DEC_RNN, ATT_RNN + MEM_D,
             ATT_RNN + MEM_D, 4 * DEC_RNN, 0);
        gemm(dec_h,    pWhd, gates_dec, nullptr, BB, 4 * DEC_RNN, DEC_RNN,
             DEC_RNN, 4 * DEC_RNN, 1);
        hipLaunchKernelGGL(lstm_cell, dim3((BB * DEC_RNN + 255) / 256), dim3(256), 0,
                           stream, gates_dec, b_dec, dec_h, dec_c, DEC_RNN);

        // projection -> mel[b, t, :]  (out row stride per batch = T_DEC*N_MEL)
        cat(dec_h, DEC_RNN, ctx, MEM_D, xcat_prj);
        gemm(xcat_prj, pWpr, out + (size_t)t * N_MEL, bproj,
             BB, N_MEL, DEC_RNN + MEM_D, DEC_RNN + MEM_D, T_DEC * N_MEL, 0);
    }
}
